// hubs_17746804867287
// MI455X (gfx1250) — matrix operand order repaired and checked by reading, not compiled, not run on hardware
//
#include <hip/hip_runtime.h>
#include <hip/hip_bf16.h>

typedef __attribute__((ext_vector_type(16))) _Float16 v16h;
typedef __attribute__((ext_vector_type(8)))  _Float16 v8h;
typedef __attribute__((ext_vector_type(4)))  _Float16 v4h;
typedef __attribute__((ext_vector_type(8)))  float    v8f;
typedef __attribute__((ext_vector_type(4)))  float    v4f;

union ABfrag { v16h v; v8h h8[2]; };

#define BLK   128      // threads per block (4 wave32)
#define ROWS  64       // batch rows per block
#define HP    272      // padded LDS h-row stride in halfs (544 B, 16B-aligned)
#define OP    132      // padded LDS out-row stride in floats (528 B, 16B-aligned)
#define H1    256
#define H2    128

// fast tanh via hardware transcendentals:
// tanh(x) = 1 - 2 / (exp2(2*log2(e)*x) + 1); saturates correctly at +/-inf
__device__ __forceinline__ float fast_tanh(float x) {
    float e = __builtin_amdgcn_exp2f(x * 2.885390082f);   // exp(2x)
    float r = __builtin_amdgcn_rcpf(e + 1.0f);
    return __builtin_fmaf(-2.0f, r, 1.0f);
}

// ---------------------------------------------------------------------------
// Prep: W2 [256][128] f32 (row-major) -> w2t [128][256] f16 (transposed) in ws
// ---------------------------------------------------------------------------
__global__ __launch_bounds__(128) void w2_transpose_f16(
    const float* __restrict__ W2, _Float16* __restrict__ w2t)
{
    const int e = (blockIdx.x * 128 + threadIdx.x) * 4;  // 0..32764, step 4
    const int k = e >> 7;          // row of W2   (0..255)
    const int n = e & 127;         // col of W2   (0..124, 4-aligned)
    const v4f v = *(const v4f*)(W2 + e);
    w2t[(size_t)(n + 0) * H1 + k] = (_Float16)v.x;
    w2t[(size_t)(n + 1) * H1 + k] = (_Float16)v.y;
    w2t[(size_t)(n + 2) * H1 + k] = (_Float16)v.z;
    w2t[(size_t)(n + 3) * H1 + k] = (_Float16)v.w;
}

// ---------------------------------------------------------------------------
// Main fused kernel: gather+relu -> f16 LDS tile -> WMMA -> tanh -> NT stores
// ---------------------------------------------------------------------------
__global__ __launch_bounds__(BLK) void hubs_fused_kernel(
    const int*   __restrict__ x,
    const float* __restrict__ W1,
    const float* __restrict__ b1,
    const _Float16* __restrict__ w2t,
    const float* __restrict__ b2,
    float* __restrict__ out)
{
    __shared__ union {
        _Float16 h[ROWS * HP];     // relu(W1[x]+b1) tile (gather + compute)
        float    o[ROWS * OP];     // output staging tile (epilogue)
    } smem;
    __shared__ int xs[ROWS];

    const int t       = threadIdx.x;
    const int rowBase = blockIdx.x * ROWS;

    if (t < ROWS) xs[t] = x[rowBase + t];
    __syncthreads();

    // ---- Gather phase: 2 rows per pass, 64 lanes/row, one float4 each ----
    {
        const int sub = t >> 6;        // which of the 2 rows this pass
        const int c4  = t & 63;        // float4 index within the 256-wide row
        const v4f bb  = *(const v4f*)(b1 + c4 * 4);

        for (int pass = 0; pass < ROWS / 2; ++pass) {
            const int r  = pass * 2 + sub;
            const int xi = xs[r];
            if (pass + 2 < ROWS / 2) {   // prefetch row needed 2 passes ahead
                const int xn = xs[(pass + 2) * 2 + sub];
                __builtin_prefetch(W1 + (size_t)xn * H1 + c4 * 4, 0, 1);
            }
            const v4f v = *(const v4f*)(W1 + (size_t)xi * H1 + c4 * 4);
            v4h hv;
            hv.x = (_Float16)fmaxf(v.x + bb.x, 0.0f);
            hv.y = (_Float16)fmaxf(v.y + bb.y, 0.0f);
            hv.z = (_Float16)fmaxf(v.z + bb.z, 0.0f);
            hv.w = (_Float16)fmaxf(v.w + bb.w, 0.0f);
            *(v4h*)(&smem.h[r * HP + c4 * 4]) = hv;   // ds_store_b64
        }
    }
    __syncthreads();

    // ---- Compute phase: each wave owns one 16-row tile ----
    const int lane = t & 31;
    const int wave = t >> 5;        // 0..3
    const int m    = lane & 15;     // M (A/D) or N (B) coordinate
    const int hi   = lane >> 4;     // upper/lower half-wave selector

    float bias2[8];
#pragma unroll
    for (int n = 0; n < 8; ++n) bias2[n] = b2[n * 16 + m];

    v8f acc[8];
    {
        v8f z = {};
#pragma unroll
        for (int n = 0; n < 8; ++n) acc[n] = z;
    }

    const _Float16* hrow = &smem.h[(wave * 16 + m) * HP];

#pragma unroll
    for (int ks = 0; ks < 8; ++ks) {
        const int k0 = ks * 32;
        // A 16x32 f16: lanes 0-15 -> K [k0..k0+7] & [k0+16..k0+23];
        //              lanes 16-31 -> +8 shift.  Two ds_load_b128.
        ABfrag a;
        a.h8[0] = *(const v8h*)(hrow + k0 + hi * 8);
        a.h8[1] = *(const v8h*)(hrow + k0 + hi * 8 + 16);
#pragma unroll
        for (int n = 0; n < 8; ++n) {
            // B 32x16 f16: lane column N = n*16+m; lanes 0-15 -> K [k0..k0+7] & [k0+16..k0+23],
            //              lanes 16-31 -> +8 shift. Two global_load_b128
            //              from L2-resident transposed W2.
            const _Float16* bp = w2t + (size_t)(n * 16 + m) * H1 + k0 + hi * 8;
            ABfrag b;
            b.h8[0] = *(const v8h*)(bp);
            b.h8[1] = *(const v8h*)(bp + 16);
            acc[n] = __builtin_amdgcn_wmma_f32_16x16x32_f16(
                false, a.v, false, b.v, (short)0, acc[n], false, false);
        }
    }

    // ---- Epilogue: tanh(acc + b2) -> LDS staging (conflict-free pattern) ----
    __syncthreads();   // hbuf reads complete on all waves; safe to alias
    {
        const int or0 = wave * 16 + hi * 8;   // D layout: VGPR j -> M=j+8*hi
#pragma unroll
        for (int n = 0; n < 8; ++n) {
#pragma unroll
            for (int j = 0; j < 8; ++j) {
                smem.o[(or0 + j) * OP + n * 16 + m] =
                    fast_tanh(acc[n][j] + bias2[n]);
            }
        }
    }
    __syncthreads();

    // ---- Coalesced non-temporal b128 stores: 512 B contiguous per wave op ----
#pragma unroll
    for (int i = 0; i < 16; ++i) {
        const int f = i * BLK + t;       // float4 id, 0..2047
        const int r = f >> 5;            // row within tile (0..63)
        const int c = (f & 31) * 4;      // col (0..124)
        const v4f val = *(const v4f*)(&smem.o[r * OP + c]);
        __builtin_nontemporal_store(
            val, (v4f*)(out + (size_t)(rowBase + r) * H2 + c));
    }
}

// ---------------------------------------------------------------------------
extern "C" void kernel_launch(void* const* d_in, const int* in_sizes, int n_in,
                              void* d_out, int out_size, void* d_ws, size_t ws_size,
                              hipStream_t stream)
{
    const int*   x  = (const int*)  d_in[0];
    const float* W1 = (const float*)d_in[1];
    const float* b1 = (const float*)d_in[2];
    const float* W2 = (const float*)d_in[3];
    const float* b2 = (const float*)d_in[4];
    float* out      = (float*)d_out;

    _Float16* w2t = (_Float16*)d_ws;   // needs 256*128*2 = 64 KB of scratch

    // one-off transpose/convert of W2 (L2-resident afterwards)
    w2_transpose_f16<<<64, 128, 0, stream>>>(W2, w2t);

    const int batch   = in_sizes[0];       // 524288
    const int nblocks = batch / ROWS;      // 8192
    hubs_fused_kernel<<<nblocks, BLK, 0, stream>>>(x, W1, b1, w2t, b2, out);
}
